// DescriptorNet_31490700214403
// MI455X (gfx1250) — matrix operand order repaired and checked by reading, not compiled, not run on hardware
//
#include <hip/hip_runtime.h>
#include <hip/hip_bf16.h>

typedef _Float16 v8h  __attribute__((ext_vector_type(8)));
typedef _Float16 v16h __attribute__((ext_vector_type(16)));
typedef float    v8f  __attribute__((ext_vector_type(8)));

#define N_NEI 128
#define M_OUT 100
#define MPP   10

__device__ __forceinline__ v16h cat16(v8h lo, v8h hi) {
    return __builtin_shufflevector(lo, hi, 0,1,2,3,4,5,6,7,8,9,10,11,12,13,14,15);
}

__global__ __launch_bounds__(128)
void descriptor_wmma_kernel(const float* __restrict__ edge_vecs,
                            const float* __restrict__ W0,  const float* __restrict__ b0,
                            const float* __restrict__ W1,  const float* __restrict__ b1,
                            const float* __restrict__ Wout,const float* __restrict__ bout,
                            float* __restrict__ out)
{
    // LDS staging (~60.4 KB of the 320 KB WGP pool)
    __shared__ __align__(16) float     env[N_NEI][4];    //  2 KB   (shat, rvec)
    __shared__ __align__(64) _Float16  h0[N_NEI][32];    //  8 KB   layer-0 act, K 25->32
    __shared__ __align__(64) _Float16  h1[N_NEI][64];    // 16 KB   layer-1 act, N 50->64
    __shared__ __align__(64) _Float16  W1t[64][32];      //  4 KB   W1^T zero-padded (n,k)
    __shared__ __align__(64) _Float16  W2t[112][64];     // 14 KB   Wout^T zero-padded (n,k)
    __shared__ float b1s[64];
    __shared__ float bos[112];
    __shared__ __align__(16) float     part[8][112][4];  // 14 KB   Si partials per (wave,half)
    __shared__ __align__(16) float     Si[112][4];       // 1.8 KB

    const int tid  = threadIdx.x;
    const int atom = blockIdx.x;
    const int lane = tid & 31;
    const int wave = tid >> 5;

    // ---------------- Phase 1: stage weights, geometry, layer 0 ----------------
    for (int idx = tid; idx < 64 * 32; idx += 128) {      // W1t[n][k] = W1[k][n]
        int n = idx >> 5, k = idx & 31;
        float w = (k < 25 && n < 50) ? W1[k * 50 + n] : 0.f;
        W1t[n][k] = (_Float16)w;
    }
    for (int idx = tid; idx < 112 * 64; idx += 128) {     // W2t[n][k] = Wout[k][n]
        int n = idx >> 6, k = idx & 63;
        float w = (k < 50 && n < 100) ? Wout[k * 100 + n] : 0.f;
        W2t[n][k] = (_Float16)w;
    }
    if (tid < 64)  b1s[tid] = (tid < 50)  ? b1[tid]   : 0.f;
    if (tid < 112) bos[tid] = (tid < 100) ? bout[tid] : 0.f;

    {   // one edge per thread: switching function + env + layer 0 (rank-1 input)
        const float* ev = edge_vecs + ((size_t)atom * N_NEI + tid) * 3;
        float x = ev[0], y = ev[1], z = ev[2];
        float r    = sqrtf(x*x + y*y + z*z);
        float rs   = fmaxf(r, 1e-8f);
        float invr = 1.f / rs;
        float u    = (rs - 2.0f) * 0.25f;                 // (r - R_CS)/(R_C - R_CS)
        float poly = u*u*u * (-6.f*u*u + 15.f*u - 10.f);
        float smid = invr * (poly + 1.f);
        float sij  = (rs < 2.f) ? invr : ((rs < 6.f) ? smid : 0.f);
        float shat = (sij - 0.05f) * 10.f;                // (sij - S_MEAN)/S_STD
        float sc   = sij * 2.f;                           // sij / COORDS_STD
        env[tid][0] = shat; env[tid][1] = x*sc; env[tid][2] = y*sc; env[tid][3] = z*sc;
        #pragma unroll
        for (int j = 0; j < 25; ++j)
            h0[tid][j] = (_Float16)fmaxf(shat * W0[j] + b0[j], 0.f);
        #pragma unroll
        for (int j = 25; j < 32; ++j)
            h0[tid][j] = (_Float16)0.f;
    }
    __syncthreads();

    // Fragment geometry (per documented 16-bit VGPR layouts)
    const int row  = lane & 15;
    const int half = lane >> 4;       // 0: lanes 0-15, 1: lanes 16-31
    const int akb  = half * 8;        // A frag: elems 0-7 = K akb.., 8-15 = K akb+16..
    const int bkb  = half * 8;        // B frag: elems 0-7 = K bkb.., 8-15 = K bkb+16..

    // ---------------- Phase 2: layer 1, (128x32) @ (32x64), ReLU ----------------
    #pragma unroll
    for (int i = 0; i < 2; ++i) {
        const int m0 = (wave * 2 + i) * 16;
        const int m  = m0 + row;
        v16h a = cat16(*(const v8h*)&h0[m][akb], *(const v8h*)&h0[m][akb + 16]);
        #pragma unroll
        for (int nt = 0; nt < 4; ++nt) {
            const int n = nt * 16 + row;
            v16h b = cat16(*(const v8h*)&W1t[n][bkb], *(const v8h*)&W1t[n][bkb + 16]);
            v8f c = {};
            c = __builtin_amdgcn_wmma_f32_16x16x32_f16(false, a, false, b,
                                                       (short)0, c, false, false);
            const float bias = b1s[n];
            #pragma unroll
            for (int j = 0; j < 8; ++j) {
                const int me = m0 + j + half * 8;         // edge row of D element j
                h1[me][n] = (_Float16)fmaxf(c[j] + bias, 0.f);
            }
        }
    }
    __syncthreads();

    // ------- Phase 3: layer 2, (128x64) @ (64x112), register-fused Si -------
    {
        const int m0a = (wave * 2 + 0) * 16;
        const int m0b = (wave * 2 + 1) * 16;
        const int ma  = m0a + row;
        const int mb  = m0b + row;
        // Hoisted A fragments: both M-tiles, both K-steps (K 0..31, 32..63)
        v16h a00 = cat16(*(const v8h*)&h1[ma][akb],      *(const v8h*)&h1[ma][akb + 16]);
        v16h a01 = cat16(*(const v8h*)&h1[ma][32 + akb], *(const v8h*)&h1[ma][32 + akb + 16]);
        v16h a10 = cat16(*(const v8h*)&h1[mb][akb],      *(const v8h*)&h1[mb][akb + 16]);
        v16h a11 = cat16(*(const v8h*)&h1[mb][32 + akb], *(const v8h*)&h1[mb][32 + akb + 16]);

        const int slot = wave * 2 + half;                 // 0..7 partial slot
        for (int nt = 0; nt < 7; ++nt) {
            const int n = nt * 16 + row;                  // output feature (padded to 112)
            v16h bb0 = cat16(*(const v8h*)&W2t[n][bkb],      *(const v8h*)&W2t[n][bkb + 16]);
            v16h bb1 = cat16(*(const v8h*)&W2t[n][32 + bkb], *(const v8h*)&W2t[n][32 + bkb + 16]);
            v8f c0 = {};
            c0 = __builtin_amdgcn_wmma_f32_16x16x32_f16(false, a00, false, bb0,
                                                        (short)0, c0, false, false);
            c0 = __builtin_amdgcn_wmma_f32_16x16x32_f16(false, a01, false, bb1,
                                                        (short)0, c0, false, false);
            v8f c1 = {};
            c1 = __builtin_amdgcn_wmma_f32_16x16x32_f16(false, a10, false, bb0,
                                                        (short)0, c1, false, false);
            c1 = __builtin_amdgcn_wmma_f32_16x16x32_f16(false, a11, false, bb1,
                                                        (short)0, c1, false, false);
            // Si[n][f] partial over this lane's 16 edge rows (no divergence:
            // padded n>=100 has zero weights+bias -> zero contribution)
            const float bias = bos[n];
            float acc0 = 0.f, acc1 = 0.f, acc2 = 0.f, acc3 = 0.f;
            #pragma unroll
            for (int j = 0; j < 8; ++j) {
                const int me = m0a + j + half * 8;
                const float g = c0[j] + bias;             // Gi[edge me][feature n]
                const float4 e = *(const float4*)&env[me][0];
                acc0 += g * e.x; acc1 += g * e.y; acc2 += g * e.z; acc3 += g * e.w;
            }
            #pragma unroll
            for (int j = 0; j < 8; ++j) {
                const int me = m0b + j + half * 8;
                const float g = c1[j] + bias;
                const float4 e = *(const float4*)&env[me][0];
                acc0 += g * e.x; acc1 += g * e.y; acc2 += g * e.z; acc3 += g * e.w;
            }
            *(float4*)&part[slot][n][0] = make_float4(acc0, acc1, acc2, acc3);
        }
    }
    __syncthreads();

    // ------- Phase 4: deterministic 8-way reduction of Si partials -------
    for (int idx = tid; idx < 112 * 4; idx += 128) {
        const int n = idx >> 2, f = idx & 3;
        float s = 0.f;
        #pragma unroll
        for (int p = 0; p < 8; ++p) s += part[p][n][f];
        Si[n][f] = s;
    }
    __syncthreads();

    // ---------------- Phase 5: Di[m][p] = <Si[m], Si[p]>, p < 10 ----------------
    float* outA = out + (size_t)atom * (M_OUT * MPP);
    for (int idx = tid; idx < M_OUT * MPP; idx += 128) {
        const int m = idx / MPP, p = idx - m * MPP;
        outA[idx] = Si[m][0]*Si[p][0] + Si[m][1]*Si[p][1]
                  + Si[m][2]*Si[p][2] + Si[m][3]*Si[p][3];
    }
}

extern "C" void kernel_launch(void* const* d_in, const int* in_sizes, int n_in,
                              void* d_out, int out_size, void* d_ws, size_t ws_size,
                              hipStream_t stream) {
    const float* edge_vecs = (const float*)d_in[0];
    const float* W0   = (const float*)d_in[1];
    const float* b0   = (const float*)d_in[2];
    const float* W1   = (const float*)d_in[3];
    const float* b1   = (const float*)d_in[4];
    const float* Wout = (const float*)d_in[5];
    const float* bout = (const float*)d_in[6];
    float* out = (float*)d_out;

    const int n_atoms = in_sizes[0] / (N_NEI * 3);   // 10000
    descriptor_wmma_kernel<<<n_atoms, 128, 0, stream>>>(
        edge_vecs, W0, b0, W1, b1, Wout, bout, out);
}
